// FactorAtt_ConvRelPosEnc_21930103014109
// MI455X (gfx1250) — compile-verified, matrix operand order checked by reading, not run on hardware
//
#include <hip/hip_runtime.h>

typedef __attribute__((ext_vector_type(4)))  __bf16 v4bf;
typedef __attribute__((ext_vector_type(8)))  __bf16 v8bf;
typedef __attribute__((ext_vector_type(16))) __bf16 v16bf;
typedef __attribute__((ext_vector_type(8)))  float  v8f;
typedef __attribute__((ext_vector_type(4)))  float  v4f;

#define DEV __device__ __forceinline__

#if __has_builtin(__builtin_amdgcn_global_load_async_to_lds_b128) && \
    __has_builtin(__builtin_amdgcn_s_wait_asynccnt)
#define CDNA5_ASYNC 1
#else
#define CDNA5_ASYNC 0
#endif

// The async builtin wants pointers to <4 x i32> in explicit address spaces:
// (global int4*, local int4*, imm offset, imm cpol)
typedef int v4i_vs __attribute__((vector_size(16)));
typedef __attribute__((address_space(1))) v4i_vs g_v4i;
typedef __attribute__((address_space(3))) v4i_vs l_v4i;

// ---------- helpers ----------
DEV __bf16 f2bf(float f) {
  union { float f; unsigned u; } x; x.f = f;
  unsigned r = x.u + 0x7FFFu + ((x.u >> 16) & 1u);   // round-to-nearest-even
  union { unsigned short s; __bf16 b; } c;
  c.s = (unsigned short)(r >> 16);
  return c.b;
}

DEV v8f zero8() {
  v8f z;
#pragma unroll
  for (int i = 0; i < 8; ++i) z[i] = 0.f;
  return z;
}

// 16-byte global -> LDS copy; async (ASYNCcnt) when the toolchain has it.
DEV void copy16(const __bf16* g, __bf16* l) {
#if CDNA5_ASYNC
  __builtin_amdgcn_global_load_async_to_lds_b128((g_v4i*)g, (l_v4i*)l, 0, 0);
#else
  *(v8bf*)l = *(const v8bf*)g;
#endif
}

DEV void wait_async_then_barrier() {
#if CDNA5_ASYNC
  __builtin_amdgcn_s_wait_asynccnt(0);
#endif
  __syncthreads();
}

// Load a 16-bit A/B^T WMMA fragment from an LDS row (ISA 16-bit 16x32 layout):
// lane half=0 holds K = k0+[0..7] and k0+16+[0..7]; half=1 holds +8 of each.
DEV v16bf ld_frag(const __bf16* row, int k0, int half) {
  v8bf lo = *(const v8bf*)(row + k0 + 8 * half);
  v8bf hi = *(const v8bf*)(row + k0 + 16 + 8 * half);
  v16bf r;
#pragma unroll
  for (int i = 0; i < 8; ++i) { r[i] = lo[i]; r[i + 8] = hi[i]; }
  return r;
}

DEV v8f wmma_bf16(v16bf a, v16bf b, v8f c) {
  return __builtin_amdgcn_wmma_f32_16x16x32_bf16(false, a, false, b, (short)0, c,
                                                 false, false);
}

// Problem constants
constexpr long QKV_PART = 32L * 8 * 1024 * 96;  // 25165824 floats each of q,k,v

// =====================================================================
// Kernel 0: streaming f32 -> bf16 conversion (one pass over x and w).
// =====================================================================
__global__ __launch_bounds__(256) void cvt_f32_bf16(const float* __restrict__ src,
                                                    __bf16* __restrict__ dst,
                                                    int n4) {
  int i = blockIdx.x * 256 + threadIdx.x;
  if (i < n4) {
    v4f v = *(const v4f*)(src + (size_t)i * 4);
    v4bf o;
#pragma unroll
    for (int j = 0; j < 4; ++j) o[j] = f2bf(v[j]);
    *(v4bf*)(dst + (size_t)i * 4) = o;
  }
}

// =====================================================================
// Kernel 1: qkv = x @ w^T, written directly into [3][B][h][N][Ch] f32.
// M=32768, N=2304, K=768. 128x128 block tile, bf16 WMMA, f32 accum,
// double-buffered async global->LDS staging.
// =====================================================================
__global__ __launch_bounds__(256) void qkv_gemm(const __bf16* __restrict__ xb,
                                                const __bf16* __restrict__ wb,
                                                float* __restrict__ qkv) {
  __shared__ __align__(16) __bf16 As[2][128 * 32];
  __shared__ __align__(16) __bf16 Bs[2][128 * 32];

  const int t = threadIdx.x;
  const int lane = t & 31;
  const int wid = __builtin_amdgcn_readfirstlane(t >> 5);   // wave-uniform
  const int half = lane >> 4, l16 = lane & 15;
  const int wm = wid & 3, wn = wid >> 2;          // 4 (M) x 2 (N) wave grid
  const int mBase = blockIdx.y * 128;
  const int nBase = blockIdx.x * 128;

  // this thread's two 16B staging chunks per 128x32 tile
  const int c0 = t, c1 = t + 256;                 // chunk ids in [0,512)
  const int r0 = c0 >> 2, o0 = (c0 & 3) * 8;
  const int r1 = c1 >> 2, o1 = (c1 & 3) * 8;

  auto stage = [&](int buf, int k0) {
    copy16(xb + (size_t)(mBase + r0) * 768 + k0 + o0, &As[buf][r0 * 32 + o0]);
    copy16(xb + (size_t)(mBase + r1) * 768 + k0 + o1, &As[buf][r1 * 32 + o1]);
    copy16(wb + (size_t)(nBase + r0) * 768 + k0 + o0, &Bs[buf][r0 * 32 + o0]);
    copy16(wb + (size_t)(nBase + r1) * 768 + k0 + o1, &Bs[buf][r1 * 32 + o1]);
  };

  v8f acc[2][4];
#pragma unroll
  for (int i = 0; i < 2; ++i)
#pragma unroll
    for (int j = 0; j < 4; ++j) acc[i][j] = zero8();

  stage(0, 0);
  wait_async_then_barrier();

  int cur = 0;
  for (int k0 = 0; k0 < 768; k0 += 32) {
    int nxt = cur ^ 1;
    if (k0 + 32 < 768) stage(nxt, k0 + 32);       // overlap with WMMA below

    v16bf af[2], bfr[4];
#pragma unroll
    for (int i = 0; i < 2; ++i)
      af[i] = ld_frag(&As[cur][(wm * 32 + i * 16 + l16) * 32], 0, half);
#pragma unroll
    for (int j = 0; j < 4; ++j)
      bfr[j] = ld_frag(&Bs[cur][(wn * 64 + j * 16 + l16) * 32], 0, half);
#pragma unroll
    for (int i = 0; i < 2; ++i)
#pragma unroll
      for (int j = 0; j < 4; ++j) acc[i][j] = wmma_bf16(af[i], bfr[j], acc[i][j]);

    wait_async_then_barrier();
    cur = nxt;
  }

  // scatter epilogue: all div/mod is wave-uniform -> SALU; lanes add a fixed offset.
  const int laneOff = half * 768 + l16;           // (half*8 rows)*96 + l16
#pragma unroll
  for (int i = 0; i < 2; ++i)
    for (int j = 0; j < 4; ++j) {
      int gnb = nBase + wn * 64 + j * 16;         // 16-aligned, never crosses 96
      int s = gnb / 768; int rem = gnb - s * 768;
      int head = rem / 96; int chib = rem - head * 96;
#pragma unroll
      for (int r = 0; r < 8; ++r) {
        int gmb = mBase + wm * 32 + i * 16 + r;   // +half*8 never wraps 1024
        int bidx = gmb >> 10; int tokb = gmb & 1023;
        size_t baseIdx = (((size_t)(s * 32 + bidx) * 8 + head) * 1024 + tokb) * 96 + chib;
        qkv[baseIdx + laneOff] = acc[i][j][r];
      }
    }
}

// =====================================================================
// Kernel 2: softmax over the token axis of k (in place, f32).
// One wave per (b,h,chi) column of 1024 tokens; 24576 columns total.
// =====================================================================
__global__ __launch_bounds__(256) void ksoftmax(float* __restrict__ kpart) {
  const int t = threadIdx.x, lane = t & 31;
  const int wid = __builtin_amdgcn_readfirstlane(t >> 5);
  const int col = blockIdx.x * 8 + wid;            // < 24576
  const int bh = col / 96, chi = col - bh * 96;
  float* p = kpart + (size_t)bh * 1024 * 96 + chi;

  float v[32];
  float m = -1e30f;
#pragma unroll
  for (int i = 0; i < 32; ++i) {
    v[i] = p[(size_t)(i * 32 + lane) * 96];
    m = fmaxf(m, v[i]);
  }
#pragma unroll
  for (int off = 16; off >= 1; off >>= 1) m = fmaxf(m, __shfl_xor(m, off, 32));
  float s = 0.f;
#pragma unroll
  for (int i = 0; i < 32; ++i) { v[i] = __expf(v[i] - m); s += v[i]; }
#pragma unroll
  for (int off = 16; off >= 1; off >>= 1) s += __shfl_xor(s, off, 32);
  float inv = 1.f / s;
#pragma unroll
  for (int i = 0; i < 32; ++i) p[(size_t)(i * 32 + lane) * 96] = v[i] * inv;
}

// =====================================================================
// Kernel 3: dd[b,h] = k_softmax^T @ v : 96x96, K=1024.
// One block (6 waves) per (b,h); stage transposed bf16 tiles in LDS.
// =====================================================================
__global__ __launch_bounds__(192) void dd_gemm(const float* __restrict__ ks,
                                               const float* __restrict__ v,
                                               float* __restrict__ dd) {
  __shared__ __align__(16) __bf16 kst[96 * 32];   // [c][n_local]
  __shared__ __align__(16) __bf16 vt[96 * 32];    // [d][n_local]
  const int t = threadIdx.x, lane = t & 31;
  const int wid = __builtin_amdgcn_readfirstlane(t >> 5);
  const int half = lane >> 4, l16 = lane & 15;
  const int bh = blockIdx.x;
  const int mo = wid * 16;                         // c-tile row for this wave
  const float* ksp = ks + (size_t)bh * 1024 * 96;
  const float* vp  = v  + (size_t)bh * 1024 * 96;

  v8f acc[6];
#pragma unroll
  for (int j = 0; j < 6; ++j) acc[j] = zero8();

  for (int n0 = 0; n0 < 1024; n0 += 32) {
    __syncthreads();
    for (int idx = t; idx < 32 * 96; idx += 192) {
      int nl = idx / 96, c = idx - nl * 96;
      kst[c * 32 + nl] = f2bf(ksp[(size_t)(n0 + nl) * 96 + c]);
      vt [c * 32 + nl] = f2bf(vp [(size_t)(n0 + nl) * 96 + c]);
    }
    __syncthreads();
    v16bf a = ld_frag(&kst[(mo + l16) * 32], 0, half);
#pragma unroll
    for (int j = 0; j < 6; ++j) {
      v16bf b = ld_frag(&vt[(j * 16 + l16) * 32], 0, half);
      acc[j] = wmma_bf16(a, b, acc[j]);
    }
  }

  float* ddp = dd + (size_t)bh * 96 * 96;
  const int laneOff = half * 768 + l16;           // (half*8)*96 + l16
#pragma unroll
  for (int j = 0; j < 6; ++j)
#pragma unroll
    for (int r = 0; r < 8; ++r)
      ddp[(mo + r) * 96 + j * 16 + laneOff] = acc[j][r];
}

// =====================================================================
// Kernel 4: out = SCALE * (q @ dd) + q * (dwconv3x3(v) + bias), fused.
// One block (8 waves) per (b,h); 128 tokens per iteration.
// =====================================================================
__global__ __launch_bounds__(256) void final_fuse(const float* __restrict__ q,
                                                  const float* __restrict__ v,
                                                  const float* __restrict__ dd,
                                                  const float* __restrict__ wc,
                                                  const float* __restrict__ bc,
                                                  float* __restrict__ out) {
  __shared__ __align__(16) __bf16 ddt[96 * 96];   // [d][c]  (B^T layout)
  __shared__ __align__(16) __bf16 qt[128 * 96];   // [tok_local][c]
  __shared__ float wl[96 * 9];
  __shared__ float bl[96];
  const int t = threadIdx.x, lane = t & 31;
  const int wid = __builtin_amdgcn_readfirstlane(t >> 5);
  const int half = lane >> 4, l16 = lane & 15;
  const int bh = blockIdx.x;
  const int bidx = bh >> 3, head = bh & 7;
  const float* qp  = q  + (size_t)bh * 1024 * 96;
  const float* vp  = v  + (size_t)bh * 1024 * 96;
  const float* ddp = dd + (size_t)bh * 96 * 96;
  const float scale = 0.102062072615966f;  // 96^-0.5

  for (int idx = t; idx < 96 * 96; idx += 256) {
    int d = idx / 96, c = idx - d * 96;
    ddt[d * 96 + c] = f2bf(ddp[c * 96 + d]);
  }
  for (int idx = t; idx < 96 * 9; idx += 256) wl[idx] = wc[(size_t)(head * 96) * 9 + idx];
  for (int idx = t; idx < 96; idx += 256) bl[idx] = bc[head * 96 + idx];

  for (int it = 0; it < 8; ++it) {
    const int tokBase = it * 128;
    __syncthreads();
    for (int idx = t; idx < 128 * 96; idx += 256) {
      int r = idx / 96, c = idx - r * 96;
      qt[r * 96 + c] = f2bf(qp[(size_t)(tokBase + r) * 96 + c]);
    }
    __syncthreads();

    const __bf16* arow = &qt[(wid * 16 + l16) * 96];
    v8f acc[6];
#pragma unroll
    for (int j = 0; j < 6; ++j) acc[j] = zero8();
#pragma unroll
    for (int k0 = 0; k0 < 96; k0 += 32) {
      v16bf a = ld_frag(arow, k0, half);
#pragma unroll
      for (int j = 0; j < 6; ++j) {
        v16bf b = ld_frag(&ddt[(j * 16 + l16) * 96], k0, half);
        acc[j] = wmma_bf16(a, b, acc[j]);
      }
    }

    // fused crpe epilogue (scalar base address + per-lane offset)
    const size_t outBase = ((size_t)bidx * 1024 + tokBase) * 768 + head * 96;
#pragma unroll
    for (int j = 0; j < 6; ++j)
#pragma unroll
      for (int r = 0; r < 8; ++r) {
        int tl  = wid * 16 + r + half * 8;
        int tok = tokBase + tl;
        int chi = j * 16 + l16;
        float qv = qp[(size_t)tok * 96 + chi];
        int y = tok >> 5, xc = tok & 31;
        float conv = bl[chi];
#pragma unroll
        for (int dy = 0; dy < 3; ++dy) {
          int yy = y + dy - 1;
          if ((unsigned)yy < 32u) {
#pragma unroll
            for (int dx = 0; dx < 3; ++dx) {
              int xx = xc + dx - 1;
              if ((unsigned)xx < 32u)
                conv += vp[(size_t)(yy * 32 + xx) * 96 + chi] * wl[chi * 9 + dy * 3 + dx];
            }
          }
        }
        out[outBase + (size_t)tl * 768 + chi] = scale * acc[j][r] + qv * conv;
      }
  }
}

// =====================================================================
extern "C" void kernel_launch(void* const* d_in, const int* in_sizes, int n_in,
                              void* d_out, int out_size, void* d_ws, size_t ws_size,
                              hipStream_t stream) {
  (void)in_sizes; (void)n_in; (void)out_size; (void)ws_size;
  const float* x  = (const float*)d_in[0];   // [32,1024,768]
  const float* wq = (const float*)d_in[1];   // [2304,768]
  const float* wc = (const float*)d_in[2];   // [768,1,3,3]
  const float* bc = (const float*)d_in[3];   // [768]
  float* out = (float*)d_out;                // [32,1024,768]

  // workspace layout (f32 qkv 288MB + dd 9MB + bf16 x 48MB + bf16 w 3.4MB)
  float* qkvF = (float*)d_ws;
  float* ddW  = qkvF + 3 * QKV_PART;
  __bf16* xb  = (__bf16*)(ddW + 256 * 96 * 96);
  __bf16* wb  = xb + 32768L * 768;
  float* qF = qkvF;
  float* kF = qkvF + QKV_PART;
  float* vF = qkvF + 2 * QKV_PART;

  const int nx4 = 32768 * 768 / 4, nw4 = 2304 * 768 / 4;
  cvt_f32_bf16<<<(nx4 + 255) / 256, 256, 0, stream>>>(x, xb, nx4);
  cvt_f32_bf16<<<(nw4 + 255) / 256, 256, 0, stream>>>(wq, wb, nw4);
  qkv_gemm<<<dim3(2304 / 128, 32768 / 128), 256, 0, stream>>>(xb, wb, qkvF);
  ksoftmax<<<(32 * 8 * 96) / 8, 256, 0, stream>>>(kF);
  dd_gemm<<<32 * 8, 192, 0, stream>>>(kF, vF, ddW);
  final_fuse<<<32 * 8, 256, 0, stream>>>(qF, vF, ddW, wc, bc, out);
}
